// Attention_14370960572643
// MI455X (gfx1250) — matrix operand order repaired and checked by reading, not compiled, not run on hardware
//
#include <hip/hip_runtime.h>

// ---------------- problem constants ----------------
#define BB   16
#define NN   100
#define HH   300
#define BIN  11
#define EE   20000
#define KTOT 311       // H + BIN
#define KPAD 320       // 10 k-steps of 32
#define NCT  19        // 19*16 = 304 >= 300 column tiles
#define JT   7         // 7*16 = 112 >= 100 row tiles
#define WSTR 328       // LDS row stride (bf16 elems) for bank spread

typedef __attribute__((ext_vector_type(16))) __bf16 v16bf;
typedef __attribute__((ext_vector_type(8)))  float  v8f;

__device__ __forceinline__ unsigned short f2bf(float f) {
    unsigned int u = __float_as_uint(f);
    u = (u + 0x7FFFu + ((u >> 16) & 1u)) >> 16;   // RNE
    return (unsigned short)u;
}

__device__ __forceinline__ float sigmoidf_(float x) {
    return 1.f / (1.f + __expf(-x));
}

// ---------------------------------------------------------------------------
// Kernel 1: fused  score = sigmoid(relu([lp|bin]·W1 + b1)·W2 + b2)  and
//           global_feats[b,i,:] = sum_j score(b,i,j) * local_feats[b,j,:]
// One wave per (b,i).  grid = 200 blocks x 256 thr (8 waves) = 1600 waves.
// ---------------------------------------------------------------------------
__global__ __launch_bounds__(256, 1)
void fused_score_global(const float* __restrict__ lf,     // [B,N,H]
                        const float* __restrict__ bfeat,  // [B,N,N,BIN]
                        const float* __restrict__ W1,     // [KTOT,H]
                        const float* __restrict__ b1,     // [H]
                        const float* __restrict__ W2,     // [H]
                        const float* __restrict__ b2,     // [1]
                        float* __restrict__ gfeat)        // ws: [B,N,H]
{
    extern __shared__ unsigned char smem[];
    unsigned short* w1t = (unsigned short*)smem;                    // [304][WSTR] bf16
    float*          b1s = (float*)(smem + 304 * WSTR * 2);          // [304]
    float*          w2s = b1s + 304;                                // [304]

    const int tid = threadIdx.x;

    // ---- stage W1 (transposed, bf16, zero-padded) + b1 + W2 into LDS ----
    for (int idx = tid; idx < KPAD * 304; idx += 256) {
        int k = idx / 304, n = idx % 304;                 // lanes sweep n: coalesced W1 reads
        float v = (k < KTOT && n < HH) ? W1[k * HH + n] : 0.f;
        w1t[n * WSTR + k] = f2bf(v);
    }
    for (int idx = tid; idx < 304; idx += 256) {
        b1s[idx] = (idx < HH) ? b1[idx] : 0.f;
        w2s[idx] = (idx < HH) ? W2[idx] : 0.f;
    }
    __syncthreads();

    const int lane = tid & 31;
    const int half = lane >> 4;        // 0: lanes 0-15, 1: lanes 16-31
    const int hl   = lane & 15;
    const int gw   = blockIdx.x * 8 + (tid >> 5);   // wave id == (b*N + i)
    if (gw >= BB * NN) return;                      // wave-uniform (never taken: 200*8==1600)

    const int b = gw / NN, i = gw % NN;
    const float* lf_b = lf + (size_t)b * NN * HH;
    const float* lfi  = lf_b + (size_t)i * HH;
    const float* bfb  = bfeat + ((size_t)b * NN + i) * NN * BIN;
    const float  b2v  = b2[0];

    float gacc[10];
#pragma unroll
    for (int c = 0; c < 10; ++c) gacc[c] = 0.f;

    for (int jt = 0; jt < JT; ++jt) {
        const int j0 = jt * 16;
        const int jr = j0 + hl;                 // this lane's A-matrix row (j)
        const bool jok = (jr < NN);
        const float* lfj = lf_b + (size_t)(jok ? jr : 0) * HH;
        const float* bfr = bfb + (size_t)(jok ? jr : 0) * BIN;

        // ---- build A fragments: 16x32 bf16 per k-step, ISA lane layout ----
        // lane(hl,half): element e -> K = s*32 + 8*half + (e<8 ? e : 8+e)
        v16bf afrag[10];
#pragma unroll
        for (int s = 0; s < 10; ++s) {
            float xv[16];
            if (s < 9) {                                    // pure local-pair region
                const int k0 = s * 32 + 8 * half;
                float4 a0 = *(const float4*)(lfi + k0);
                float4 a1 = *(const float4*)(lfi + k0 + 4);
                float4 c0 = *(const float4*)(lfj + k0);
                float4 c1 = *(const float4*)(lfj + k0 + 4);
                const int k1 = k0 + 16;
                float4 a2 = *(const float4*)(lfi + k1);
                float4 a3 = *(const float4*)(lfi + k1 + 4);
                float4 c2 = *(const float4*)(lfj + k1);
                float4 c3 = *(const float4*)(lfj + k1 + 4);
                xv[0]=a0.x+c0.x; xv[1]=a0.y+c0.y; xv[2]=a0.z+c0.z; xv[3]=a0.w+c0.w;
                xv[4]=a1.x+c1.x; xv[5]=a1.y+c1.y; xv[6]=a1.z+c1.z; xv[7]=a1.w+c1.w;
                xv[8]=a2.x+c2.x; xv[9]=a2.y+c2.y; xv[10]=a2.z+c2.z; xv[11]=a2.w+c2.w;
                xv[12]=a3.x+c3.x; xv[13]=a3.y+c3.y; xv[14]=a3.z+c3.z; xv[15]=a3.w+c3.w;
            } else {                                        // tail: lf / binary / zero-pad
#pragma unroll
                for (int e = 0; e < 16; ++e) {
                    int k = 288 + 8 * half + ((e < 8) ? e : 8 + e);
                    float v;
                    if (k < HH)          v = lfi[k] + lfj[k];
                    else if (k < KTOT)   v = bfr[k - HH];
                    else                 v = 0.f;
                    xv[e] = v;
                }
            }
            union { v16bf v; unsigned short h[16]; } af;
#pragma unroll
            for (int e = 0; e < 16; ++e) af.h[e] = f2bf(xv[e]);
            afrag[s] = af.v;
        }

        // ---- GEMM over 19 column tiles, fused relu/+b1/ *W2 reduction ----
        float p[8];
#pragma unroll
        for (int r = 0; r < 8; ++r) p[r] = 0.f;

        for (int ct = 0; ct < NCT; ++ct) {
            const int n = ct * 16 + hl;                    // this lane's output column
            const unsigned short* wrow = w1t + n * WSTR;
            v8f acc = {0.f,0.f,0.f,0.f,0.f,0.f,0.f,0.f};
#pragma unroll
            for (int s = 0; s < 10; ++s) {
                // B 32x16 bf16: lane n=hl, K = 8*half + (e<8 ? e : 8+e)  (two ds_load_b128)
                union { v16bf v; uint4 u[2]; } bfg;
                const unsigned short* bp = wrow + s * 32 + 8 * half;
                bfg.u[0] = *(const uint4*)(bp);
                bfg.u[1] = *(const uint4*)(bp + 16);
                acc = __builtin_amdgcn_wmma_f32_16x16x32_bf16(
                        false, afrag[s], false, bfg.v, (short)0, acc, false, false);
            }
            const float bias = b1s[n], w2v = w2s[n];
#pragma unroll
            for (int r = 0; r < 8; ++r) {
                float a = acc[r] + bias;
                a = a > 0.f ? a : 0.f;                     // relu
                p[r] += a * w2v;                           // ·W2 partial over n
            }
        }

        // ---- reduce over the 16 lanes sharing each row m = r + 8*half ----
        float sc[8];
#pragma unroll
        for (int r = 0; r < 8; ++r) {
            float t = p[r];
            t += __shfl_xor(t, 1);
            t += __shfl_xor(t, 2);
            t += __shfl_xor(t, 4);
            t += __shfl_xor(t, 8);
            const int m = r + 8 * half;
            sc[r] = (j0 + m < NN) ? sigmoidf_(t + b2v) : 0.f;   // mask padded j rows
        }

        // ---- global_feats accumulation: gacc[k] += score_t * lf[b, j0+t, k] ----
#pragma unroll
        for (int t = 0; t < 16; ++t) {
            const int j = j0 + t;
            if (j < NN) {                                  // wave-uniform
                const float st = (t < 8) ? __shfl(sc[t], 0) : __shfl(sc[t - 8], 16);
                const float* lr = lf_b + (size_t)j * HH;
#pragma unroll
                for (int c = 0; c < 10; ++c) {
                    const int k = c * 32 + lane;
                    if (k < HH) gacc[c] += st * lr[k];
                }
            }
        }
    }

    float* gr = gfeat + (size_t)gw * HH;
#pragma unroll
    for (int c = 0; c < 10; ++c) {
        const int k = c * 32 + lane;
        if (k < HH) gr[k] = gacc[c];
    }
}

// ---------------------------------------------------------------------------
// Kernel 2: sparse gather.  out_local[e,:] = lf[b,ii,:]+lf[b,jj,:]
//           out_global[e,:] = G[b,ii,:]+G[b,jj,:]
// ---------------------------------------------------------------------------
__global__ __launch_bounds__(320)
void gather_pairs(const float* __restrict__ lf,
                  const float* __restrict__ gfeat,
                  const long long* __restrict__ sidx,     // int64 [E,3]
                  float* __restrict__ out_local,
                  float* __restrict__ out_global)
{
    const int e = blockIdx.x;
    const int k = threadIdx.x;
    const long long bq = sidx[3 * e + 0];
    const long long iq = sidx[3 * e + 1];
    const long long jq = sidx[3 * e + 2];
    if (k < HH) {
        const size_t ri = ((size_t)bq * NN + (size_t)iq) * HH;
        const size_t rj = ((size_t)bq * NN + (size_t)jq) * HH;
        out_local [(size_t)e * HH + k] = lf[ri + k]    + lf[rj + k];
        out_global[(size_t)e * HH + k] = gfeat[ri + k] + gfeat[rj + k];
    }
}

// ---------------------------------------------------------------------------
extern "C" void kernel_launch(void* const* d_in, const int* in_sizes, int n_in,
                              void* d_out, int out_size, void* d_ws, size_t ws_size,
                              hipStream_t stream)
{
    const float*      lf    = (const float*)d_in[0];       // local_feats  [B,N,H]
    const float*      bfeat = (const float*)d_in[1];       // binary_feats [B,N,N,BIN]
    const long long*  sidx  = (const long long*)d_in[2];   // sparse_idx   [E,3] int64
    const float*      W1    = (const float*)d_in[3];
    const float*      b1    = (const float*)d_in[4];
    const float*      W2    = (const float*)d_in[5];
    const float*      b2    = (const float*)d_in[6];

    float* out   = (float*)d_out;
    float* gfeat = (float*)d_ws;                           // B*N*H f32 = 1.92 MB

    const size_t lds_bytes = (size_t)304 * WSTR * 2 + 2 * 304 * 4;  // ~202 KB (WGP LDS = 320 KB)

    fused_score_global<<<200, 256, lds_bytes, stream>>>(lf, bfeat, W1, b1, W2, b2, gfeat);
    gather_pairs<<<EE, 320, 0, stream>>>(lf, gfeat, sidx,
                                         out, out + (size_t)EE * HH);
}
